// EPMoE_532575945216
// MI455X (gfx1250) — matrix operand order repaired and checked by reading, not compiled, not run on hardware
//
#include <hip/hip_runtime.h>

#define TOKENS 4096
#define HIDDEN 1024
#define INTER  2048
#define NEXP   8
#define TOPK   2
#define NROWS  (TOKENS * TOPK)   // 8192

#define BM 128
#define BN 64
#define BK 32
#define GEMM_THREADS 256         // 8 waves (wave32), 4x2 grid of 32x32 wave tiles
#define LDA 40                   // padded LDS stride (bf16 elems), 80B -> 16B aligned
#define LDB 40

typedef __attribute__((ext_vector_type(16))) __bf16 bf16x16;
typedef __attribute__((ext_vector_type(8)))  float  f32x8;

// ---------- bf16 helpers ----------
__device__ __forceinline__ unsigned short f2bf(float f) {   // RNE (cold paths)
  unsigned int u = __float_as_uint(f);
  u += 0x7FFFu + ((u >> 16) & 1u);
  return (unsigned short)(u >> 16);
}
__device__ __forceinline__ float bf2f(unsigned short h) {
  return __uint_as_float(((unsigned int)h) << 16);
}
// hot path: pack two floats -> 2x bf16 dword (lo=a, hi=b), round-half-up + v_perm
__device__ __forceinline__ unsigned pack2bf(float a, float b) {
  unsigned ua = __float_as_uint(a) + 0x8000u;
  unsigned ub = __float_as_uint(b) + 0x8000u;
#if __has_builtin(__builtin_amdgcn_perm)
  return __builtin_amdgcn_perm(ub, ua, 0x07060302u);  // {ub[31:16], ua[31:16]}
#else
  return (ua >> 16) | (ub & 0xFFFF0000u);
#endif
}

// ---------- async global->LDS copy (CDNA5), tracked by ASYNCcnt ----------
__device__ __forceinline__ void async_copy_b128x2(unsigned lds_off, const void* gptr) {
  asm volatile("global_load_async_to_lds_b128 %0, %1, off\n\t"
               "global_load_async_to_lds_b128 %0, %1, off offset:16"
               :: "v"(lds_off), "v"(gptr) : "memory");
}
__device__ __forceinline__ void wg_sync_async() {
  asm volatile("s_wait_asynccnt 0x0" ::: "memory");
  __syncthreads();
}

// ---------- WMMA fragment loaders (LDS, bf16) ----------
union FragU { uint4 q[2]; bf16x16 v; };

__device__ __forceinline__ bf16x16 load_a(const unsigned short* s, int m0, int lane) {
  const unsigned short* p = s + (m0 + (lane & 15)) * LDA + ((lane >> 4) << 3);
  FragU f;
  f.q[0] = *(const uint4*)(p);
  f.q[1] = *(const uint4*)(p + 16);
  return f.v;
}
__device__ __forceinline__ bf16x16 load_b(const unsigned short* s, int n0l, int lane) {
  const unsigned short* p = s + (n0l + (lane & 15)) * LDB + ((lane >> 4) << 3);
  FragU f;
  f.q[0] = *(const uint4*)(p);
  f.q[1] = *(const uint4*)(p + 16);
  return f.v;
}
__device__ __forceinline__ f32x8 wmma_bf16(bf16x16 a, bf16x16 b, f32x8 c) {
  return __builtin_amdgcn_wmma_f32_16x16x32_bf16(false, a, false, b, (short)0, c, false, false);
}

// ---------- routing ----------
__global__ void init_kernel(int* counts, int* cursors) {
  int t = threadIdx.x;
  if (t < NEXP) { counts[t] = 0; cursors[t] = 0; }
}

__global__ void router_kernel(const float* __restrict__ logits, int* __restrict__ expert_ids,
                              float* __restrict__ wts, int* __restrict__ counts) {
  int t = blockIdx.x * blockDim.x + threadIdx.x;
  if (t >= TOKENS) return;
  float l[NEXP];
#pragma unroll
  for (int e = 0; e < NEXP; ++e) l[e] = logits[t * NEXP + e];
  int i0 = 0; float v0 = l[0];
#pragma unroll
  for (int e = 1; e < NEXP; ++e) if (l[e] > v0) { v0 = l[e]; i0 = e; }   // first max (stable)
  int i1 = -1; float v1 = -3.4e38f;
#pragma unroll
  for (int e = 0; e < NEXP; ++e) if (e != i0 && l[e] > v1) { v1 = l[e]; i1 = e; }
  float e1 = __expf(v1 - v0);
  float inv = 1.0f / (1.0f + e1);
  wts[2 * t + 0] = inv;
  wts[2 * t + 1] = e1 * inv;
  expert_ids[2 * t + 0] = i0;
  expert_ids[2 * t + 1] = i1;
  atomicAdd(&counts[i0], 1);
  atomicAdd(&counts[i1], 1);
}

__global__ void scan_kernel(const int* __restrict__ counts, int* __restrict__ offsets) {
  if (threadIdx.x == 0) {
    int a = 0;
    for (int e = 0; e < NEXP; ++e) { offsets[e] = a; a += counts[e]; }
  }
}

__global__ void scatter_kernel(const int* __restrict__ expert_ids, const int* __restrict__ offsets,
                               int* __restrict__ cursors, int* __restrict__ row_src,
                               int* __restrict__ row_of) {
  int s = blockIdx.x * blockDim.x + threadIdx.x;
  if (s >= NROWS) return;
  int e = expert_ids[s];
  int pos = offsets[e] + atomicAdd(&cursors[e], 1);
  row_src[pos] = s;
  row_of[s] = pos;
}

__global__ void gather_kernel(const float* __restrict__ inputs, const int* __restrict__ row_src,
                              unsigned short* __restrict__ xp) {
  int pos = blockIdx.x;
  int t = row_src[pos] >> 1;
  int i = threadIdx.x * 4;
  float4 f = *(const float4*)(inputs + (size_t)t * HIDDEN + i);
  ushort4 o;
  o.x = f2bf(f.x); o.y = f2bf(f.y); o.z = f2bf(f.z); o.w = f2bf(f.w);
  *(ushort4*)(xp + (size_t)pos * HIDDEN + i) = o;
}

// ---------- GEMM 1 (fused): act = silu(xp@wi0) * (xp@wi1), bf16 out ----------
__global__ __launch_bounds__(GEMM_THREADS)
void gemm1_kernel(const unsigned short* __restrict__ xp, const float* __restrict__ wi0,
                  const float* __restrict__ wi1, unsigned short* __restrict__ act,
                  const int* __restrict__ counts, const int* __restrict__ offsets) {
  const int e   = blockIdx.z;
  const int cnt = counts[e];
  const int rt  = blockIdx.y;
  if (rt * BM >= cnt) return;
  const int row_base   = offsets[e] + rt * BM;
  const int rows_valid = min(BM, cnt - rt * BM);
  const int n0 = blockIdx.x * BN;

  const float* B0 = wi0 + (size_t)e * HIDDEN * INTER;
  const float* B1 = wi1 + (size_t)e * HIDDEN * INTER;

  __shared__ unsigned short sA[2][BM * LDA];
  __shared__ unsigned short sB0[2][BN * LDB];   // transposed [n][k]
  __shared__ unsigned short sB1[2][BN * LDB];

  const int tid  = threadIdx.x;
  const int lane = tid & 31;
  const int wave = tid >> 5;                    // 0..7
  const int wm = (wave >> 1) * 32;              // 0,32,64,96
  const int wn = (wave & 1) * 32;

  // ---- loop-invariant per-thread addressing ----
  // A: thread copies 32B of row (tid>>1), half (tid&1); 256 threads cover 128x32 bf16
  const int arow  = tid >> 1;
  const int ahalf = (tid & 1) << 4;
  int grow = row_base + arow; if (grow >= NROWS) grow = NROWS - 1;
  const unsigned short* aSrc = xp + (size_t)grow * HIDDEN + ahalf;
  const int aDstOff = arow * LDA + ahalf;
  // B: thread owns cols n4..n4+3 and k-pair {bk,bk+1}; 256 threads cover 32x64
  const int n4 = (tid & 15) << 2;
  const int bk = ((tid >> 4) & 15) << 1;
  const float* b0Src = B0 + (size_t)bk * INTER + n0 + n4;
  const float* b1Src = B1 + (size_t)bk * INTER + n0 + n4;

  auto load_tile = [&](int buf, int kk) {
    async_copy_b128x2((unsigned)(size_t)(&sA[buf][aDstOff]), aSrc + kk);
    const float* p0 = b0Src + (size_t)kk * INTER;
    float4 r0 = *(const float4*)(p0);
    float4 r1 = *(const float4*)(p0 + INTER);
    unsigned short* d0 = &sB0[buf][n4 * LDB + bk];
    *(unsigned*)(d0 + 0 * LDB) = pack2bf(r0.x, r1.x);
    *(unsigned*)(d0 + 1 * LDB) = pack2bf(r0.y, r1.y);
    *(unsigned*)(d0 + 2 * LDB) = pack2bf(r0.z, r1.z);
    *(unsigned*)(d0 + 3 * LDB) = pack2bf(r0.w, r1.w);
    const float* p1 = b1Src + (size_t)kk * INTER;
    float4 q0 = *(const float4*)(p1);
    float4 q1 = *(const float4*)(p1 + INTER);
    unsigned short* d1 = &sB1[buf][n4 * LDB + bk];
    *(unsigned*)(d1 + 0 * LDB) = pack2bf(q0.x, q1.x);
    *(unsigned*)(d1 + 1 * LDB) = pack2bf(q0.y, q1.y);
    *(unsigned*)(d1 + 2 * LDB) = pack2bf(q0.z, q1.z);
    *(unsigned*)(d1 + 3 * LDB) = pack2bf(q0.w, q1.w);
  };

  f32x8 acc0[2][2] = {};
  f32x8 acc1[2][2] = {};

  const int nk = HIDDEN / BK;
  load_tile(0, 0);
  wg_sync_async();

  for (int i = 0; i < nk; ++i) {
    const int cur = i & 1;
    if (i + 1 < nk) load_tile(1 - cur, (i + 1) * BK);
    if (i + 2 < nk) {  // pull tile i+2 weights toward cache (global_prefetch_b8)
      size_t pidx = (size_t)((i + 2) * BK + (tid >> 3)) * INTER + n0 + ((tid & 7) << 3);
      __builtin_prefetch(B0 + pidx, 0, 1);
      __builtin_prefetch(B1 + pidx, 0, 1);
    }

    bf16x16 a0  = load_a(sA[cur], wm + 0, lane);
    bf16x16 a1  = load_a(sA[cur], wm + 16, lane);
    bf16x16 b00 = load_b(sB0[cur], wn + 0, lane);
    bf16x16 b01 = load_b(sB0[cur], wn + 16, lane);
    bf16x16 b10 = load_b(sB1[cur], wn + 0, lane);
    bf16x16 b11 = load_b(sB1[cur], wn + 16, lane);

    acc0[0][0] = wmma_bf16(a0, b00, acc0[0][0]);
    acc0[0][1] = wmma_bf16(a0, b01, acc0[0][1]);
    acc0[1][0] = wmma_bf16(a1, b00, acc0[1][0]);
    acc0[1][1] = wmma_bf16(a1, b01, acc0[1][1]);
    acc1[0][0] = wmma_bf16(a0, b10, acc1[0][0]);
    acc1[0][1] = wmma_bf16(a0, b11, acc1[0][1]);
    acc1[1][0] = wmma_bf16(a1, b10, acc1[1][0]);
    acc1[1][1] = wmma_bf16(a1, b11, acc1[1][1]);

    wg_sync_async();
  }

  const int nl = lane & 15, h = lane >> 4;
#pragma unroll
  for (int im = 0; im < 2; ++im)
#pragma unroll
    for (int jn = 0; jn < 2; ++jn) {
      f32x8 h0 = acc0[im][jn];
      f32x8 h1 = acc1[im][jn];
#pragma unroll
      for (int r = 0; r < 8; ++r) {
        int m = wm + im * 16 + h * 8 + r;
        if (m < rows_valid) {
          float g  = h0[r];
          float sv = g / (1.0f + __expf(-g));     // silu
          act[(size_t)(row_base + m) * INTER + n0 + wn + jn * 16 + nl] = f2bf(sv * h1[r]);
        }
      }
    }
}

// ---------- GEMM 2: y = act @ wo, bf16 out ----------
__global__ __launch_bounds__(GEMM_THREADS)
void gemm2_kernel(const unsigned short* __restrict__ actb, const float* __restrict__ wo,
                  unsigned short* __restrict__ y, const int* __restrict__ counts,
                  const int* __restrict__ offsets) {
  const int e   = blockIdx.z;
  const int cnt = counts[e];
  const int rt  = blockIdx.y;
  if (rt * BM >= cnt) return;
  const int row_base   = offsets[e] + rt * BM;
  const int rows_valid = min(BM, cnt - rt * BM);
  const int n0 = blockIdx.x * BN;

  const float* B = wo + (size_t)e * INTER * HIDDEN;

  __shared__ unsigned short sA[2][BM * LDA];
  __shared__ unsigned short sB[2][BN * LDB];

  const int tid  = threadIdx.x;
  const int lane = tid & 31;
  const int wave = tid >> 5;
  const int wm = (wave >> 1) * 32;
  const int wn = (wave & 1) * 32;

  const int arow  = tid >> 1;
  const int ahalf = (tid & 1) << 4;
  int grow = row_base + arow; if (grow >= NROWS) grow = NROWS - 1;
  const unsigned short* aSrc = actb + (size_t)grow * INTER + ahalf;
  const int aDstOff = arow * LDA + ahalf;
  const int n4 = (tid & 15) << 2;
  const int bk = ((tid >> 4) & 15) << 1;
  const float* bSrc = B + (size_t)bk * HIDDEN + n0 + n4;

  auto load_tile = [&](int buf, int kk) {
    async_copy_b128x2((unsigned)(size_t)(&sA[buf][aDstOff]), aSrc + kk);
    const float* p = bSrc + (size_t)kk * HIDDEN;
    float4 r0 = *(const float4*)(p);
    float4 r1 = *(const float4*)(p + HIDDEN);
    unsigned short* d = &sB[buf][n4 * LDB + bk];
    *(unsigned*)(d + 0 * LDB) = pack2bf(r0.x, r1.x);
    *(unsigned*)(d + 1 * LDB) = pack2bf(r0.y, r1.y);
    *(unsigned*)(d + 2 * LDB) = pack2bf(r0.z, r1.z);
    *(unsigned*)(d + 3 * LDB) = pack2bf(r0.w, r1.w);
  };

  f32x8 acc[2][2] = {};

  const int nk = INTER / BK;
  load_tile(0, 0);
  wg_sync_async();

  for (int i = 0; i < nk; ++i) {
    const int cur = i & 1;
    if (i + 1 < nk) load_tile(1 - cur, (i + 1) * BK);
    if (i + 2 < nk) {
      size_t pidx = (size_t)((i + 2) * BK + (tid >> 3)) * HIDDEN + n0 + ((tid & 7) << 3);
      __builtin_prefetch(B + pidx, 0, 1);
    }

    bf16x16 a0 = load_a(sA[cur], wm + 0, lane);
    bf16x16 a1 = load_a(sA[cur], wm + 16, lane);
    bf16x16 b0 = load_b(sB[cur], wn + 0, lane);
    bf16x16 b1 = load_b(sB[cur], wn + 16, lane);

    acc[0][0] = wmma_bf16(a0, b0, acc[0][0]);
    acc[0][1] = wmma_bf16(a0, b1, acc[0][1]);
    acc[1][0] = wmma_bf16(a1, b0, acc[1][0]);
    acc[1][1] = wmma_bf16(a1, b1, acc[1][1]);

    wg_sync_async();
  }

  const int nl = lane & 15, h = lane >> 4;
#pragma unroll
  for (int im = 0; im < 2; ++im)
#pragma unroll
    for (int jn = 0; jn < 2; ++jn) {
      f32x8 d = acc[im][jn];
#pragma unroll
      for (int r = 0; r < 8; ++r) {
        int m = wm + im * 16 + h * 8 + r;
        if (m < rows_valid)
          y[(size_t)(row_base + m) * HIDDEN + n0 + wn + jn * 16 + nl] = f2bf(d[r]);
      }
    }
}

// ---------- combine: out[t] = w0*y[row(t,0)] + w1*y[row(t,1)] ----------
__global__ void combine_kernel(const unsigned short* __restrict__ y, const int* __restrict__ row_of,
                               const float* __restrict__ wts, float* __restrict__ out) {
  int t = blockIdx.x;
  float w0 = wts[2 * t + 0];
  float w1 = wts[2 * t + 1];
  size_t r0 = (size_t)row_of[2 * t + 0];
  size_t r1 = (size_t)row_of[2 * t + 1];
  int i = threadIdx.x * 4;
  ushort4 a = *(const ushort4*)(y + r0 * HIDDEN + i);
  ushort4 b = *(const ushort4*)(y + r1 * HIDDEN + i);
  float4 o;
  o.x = w0 * bf2f(a.x) + w1 * bf2f(b.x);
  o.y = w0 * bf2f(a.y) + w1 * bf2f(b.y);
  o.z = w0 * bf2f(a.z) + w1 * bf2f(b.z);
  o.w = w0 * bf2f(a.w) + w1 * bf2f(b.w);
  *(float4*)(out + (size_t)t * HIDDEN + i) = o;
}

extern "C" void kernel_launch(void* const* d_in, const int* in_sizes, int n_in,
                              void* d_out, int out_size, void* d_ws, size_t ws_size,
                              hipStream_t stream) {
  const float* inputs = (const float*)d_in[0];
  const float* logits = (const float*)d_in[1];
  const float* wi0    = (const float*)d_in[2];
  const float* wi1    = (const float*)d_in[3];
  const float* wo     = (const float*)d_in[4];
  float* out = (float*)d_out;

  char* ws = (char*)d_ws;
  size_t off = 0;
  auto alloc = [&](size_t bytes) -> void* {
    void* p = ws + off;
    off += (bytes + 255) & ~(size_t)255;
    return p;
  };
  int*   counts     = (int*)alloc(NEXP * sizeof(int));
  int*   cursors    = (int*)alloc(NEXP * sizeof(int));
  int*   offsets    = (int*)alloc(NEXP * sizeof(int));
  int*   expert_ids = (int*)alloc(NROWS * sizeof(int));
  float* wts        = (float*)alloc(NROWS * sizeof(float));
  int*   row_src    = (int*)alloc(NROWS * sizeof(int));
  int*   row_of     = (int*)alloc(NROWS * sizeof(int));
  unsigned short* xp   = (unsigned short*)alloc((size_t)NROWS * HIDDEN * 2);
  unsigned short* actb = (unsigned short*)alloc((size_t)NROWS * INTER * 2);
  unsigned short* yb   = (unsigned short*)alloc((size_t)NROWS * HIDDEN * 2);
  (void)in_sizes; (void)n_in; (void)out_size; (void)ws_size;

  init_kernel<<<1, 32, 0, stream>>>(counts, cursors);
  router_kernel<<<TOKENS / 256, 256, 0, stream>>>(logits, expert_ids, wts, counts);
  scan_kernel<<<1, 1, 0, stream>>>(counts, offsets);
  scatter_kernel<<<NROWS / 256, 256, 0, stream>>>(expert_ids, offsets, cursors, row_src, row_of);
  gather_kernel<<<NROWS, 256, 0, stream>>>(inputs, row_src, xp);
  gemm1_kernel<<<dim3(INTER / BN, NROWS / BM, NEXP), GEMM_THREADS, 0, stream>>>(
      xp, wi0, wi1, actb, counts, offsets);
  gemm2_kernel<<<dim3(HIDDEN / BN, NROWS / BM, NEXP), GEMM_THREADS, 0, stream>>>(
      actb, wo, yb, counts, offsets);
  combine_kernel<<<TOKENS, 256, 0, stream>>>(yb, row_of, wts, out);
}
